// SessionGAT_79293686219352
// MI455X (gfx1250) — compile-verified, matrix operand order checked by reading, not run on hardware
//
#include <hip/hip_runtime.h>
#include <stdint.h>

// Problem constants (match reference).
#define NB   16384      // nodes
#define ND   128        // feature dim
#define NH   4          // heads
#define NC   64         // channels per head
#define NHC  256        // NH*NC
#define NK   16         // top-k
#define NE   (NB*(NK+1))// edges incl. self loops
#define NT   (NB/16)    // key tiles in similarity pass
#define NEG_SLOPE 0.2f

typedef __attribute__((ext_vector_type(16))) __bf16 v16bf;
typedef __attribute__((ext_vector_type(8)))  float  v8f;

// ---------- helpers ----------

__device__ __forceinline__ unsigned short f2bf(float f) {
    unsigned u = __float_as_uint(f);
    unsigned r = (u + 0x7FFFu + ((u >> 16) & 1u)) >> 16;   // RNE
    return (unsigned short)r;
}

// order-preserving float<->int map for atomicMax on signed ints
__device__ __forceinline__ int ford(float f) {
    int i = __float_as_int(f);
    return (i >= 0) ? i : (i ^ 0x7FFFFFFF);
}
__device__ __forceinline__ float fordinv(int i) {
    return __int_as_float((i >= 0) ? i : (i ^ 0x7FFFFFFF));
}

union BfFrag { v16bf v; uint4 q[2]; };

// Load a 16x32 bf16 A-style WMMA fragment from row-major [rows][ld] bf16.
// lane L<16: row = row0+L,   K = kbase+{0..7, 16..23}
// lane L>=16: row = row0+L-16, K = kbase+{8..15, 24..31}
__device__ __forceinline__ v16bf load_frag16x32(const unsigned short* base,
                                                int row0, int ld, int kbase, int lane) {
    int r  = row0 + (lane & 15);
    int kb = kbase + ((lane >> 4) << 3);
    const uint4* p = reinterpret_cast<const uint4*>(base + r * ld + kb);
    BfFrag f;
    f.q[0] = p[0];   // halves 0..7   (K = kb..kb+7)
    f.q[1] = p[2];   // halves 8..15  (K = kb+16..kb+23)
    return f.v;
}

__device__ __forceinline__ v8f wmma_bf16(v16bf a, v16bf b, v8f c) {
    return __builtin_amdgcn_wmma_f32_16x16x32_bf16(false, a, false, b,
                                                   (short)0, c, false, false);
}

// Async DMA: 16 bytes per lane, global -> LDS. Tracked with ASYNCcnt.
// lds_off: LDS byte offset (low 32 bits of generic shared pointer).
__device__ __forceinline__ void async_ld16(unsigned int lds_off, const void* gaddr) {
    asm volatile("global_load_async_to_lds_b128 %0, %1, off"
                 :: "v"(lds_off), "v"(gaddr) : "memory");
}

// ---------- kernels ----------

__global__ __launch_bounds__(256) void k_cvt_x(const float* __restrict__ x,
                                               unsigned short* __restrict__ xb) {
    int i = blockIdx.x * 256 + threadIdx.x;
    if (i < NB * ND) xb[i] = f2bf(x[i]);
}

// W [128][256] row-major -> Wt [256][128] bf16 (column-major of W) for B fragments.
__global__ __launch_bounds__(256) void k_cvt_w(const float* __restrict__ wl,
                                               const float* __restrict__ wr,
                                               unsigned short* __restrict__ wlt,
                                               unsigned short* __restrict__ wrt) {
    int i = blockIdx.x * 256 + threadIdx.x;
    if (i >= ND * NHC) return;
    int k = i >> 8, n = i & 255;
    wlt[n * ND + k] = f2bf(wl[i]);
    wrt[n * ND + k] = f2bf(wr[i]);
}

// x_l = x@W_l + b_l ; x_r = x@W_r + b_r. One wave per 16x16 output tile, both mats.
__global__ __launch_bounds__(256) void k_linear(const unsigned short* __restrict__ xb,
                                                const unsigned short* __restrict__ wlt,
                                                const unsigned short* __restrict__ wrt,
                                                const float* __restrict__ bl,
                                                const float* __restrict__ br,
                                                float* __restrict__ xl,
                                                float* __restrict__ xr) {
    int wid  = (blockIdx.x * 256 + threadIdx.x) >> 5;  // global wave id
    int lane = threadIdx.x & 31;
    int tm = wid >> 4;   // 0..1023 (row tile)
    int tn = wid & 15;   // 0..15   (col tile)

    v16bf a[4];
#pragma unroll
    for (int c = 0; c < 4; ++c)
        a[c] = load_frag16x32(xb, tm * 16, ND, c * 32, lane);

    int col   = tn * 16 + (lane & 15);
    int rbase = tm * 16 + ((lane >> 4) << 3);

#pragma unroll
    for (int s = 0; s < 2; ++s) {
        const unsigned short* wt = s ? wrt : wlt;
        const float* bias        = s ? br  : bl;
        float* out               = s ? xr  : xl;
        v8f acc = {};
#pragma unroll
        for (int c = 0; c < 4; ++c) {
            v16bf b = load_frag16x32(wt, tn * 16, ND, c * 32, lane);
            acc = wmma_bf16(a[c], b, acc);
        }
        float bv = bias[col];
#pragma unroll
        for (int v = 0; v < 8; ++v)
            out[(rbase + v) * NHC + col] = acc[v] + bv;
    }
}

// Fused x@x^T similarity + per-row top-16 indices. One wave owns 32 query rows,
// each lane owns one row's top-k state. Key tiles are staged in LDS once per
// workgroup via async global->LDS DMA (double buffered); all 8 waves read their
// B fragments from LDS. Symmetric GEMM: B fragment uses the A-fragment layout.
__global__ __launch_bounds__(256) void k_topk(const unsigned short* __restrict__ xb,
                                              int* __restrict__ topk) {
    __shared__ unsigned short kbuf[2][16 * ND];  // 2 x 4KB key tiles
    __shared__ float stile[8][32][17];           // per-wave 32x16 score tile, padded
    int tid    = threadIdx.x;
    int wlocal = tid >> 5;
    int lane   = tid & 31;
    int rowbase = (blockIdx.x * 8 + wlocal) * 32;

    unsigned int kb_off[2];
    kb_off[0] = (unsigned int)(uintptr_t)&kbuf[0][0];
    kb_off[1] = (unsigned int)(uintptr_t)&kbuf[1][0];
    const char* gx = (const char*)xb;

    // Prologue: stage key tile 0 (256 threads x 16B = 4KB).
    async_ld16(kb_off[0] + tid * 16, gx + (size_t)tid * 16);

    v16bf a[8];
#pragma unroll
    for (int t = 0; t < 2; ++t)
#pragma unroll
        for (int c = 0; c < 4; ++c)
            a[t * 4 + c] = load_frag16x32(xb, rowbase + t * 16, ND, c * 32, lane);

    float vals[NK]; int ids[NK];
#pragma unroll
    for (int i = 0; i < NK; ++i) { vals[i] = -3.0e38f; ids[i] = 0; }
    float minval = -3.0e38f; int minpos = 0;

    int rloc = lane & 15;              // key row within tile for B fragment
    int koff = (lane >> 4) << 3;       // 0 or 8

    for (int kt = 0; kt < NT; ++kt) {
        int buf = kt & 1;
        // Kick off DMA for the next tile, then retire the current one.
        if (kt + 1 < NT) {
            async_ld16(kb_off[buf ^ 1] + tid * 16,
                       gx + (size_t)(kt + 1) * (16 * ND * 2) + tid * 16);
            asm volatile("s_wait_asynccnt 0x1" ::: "memory");
        } else {
            asm volatile("s_wait_asynccnt 0x0" ::: "memory");
        }
        __syncthreads();   // tile kt visible to all waves; prev stile consumed

        // B fragments from LDS (ds_load_b128 pairs).
        v16bf b[4];
#pragma unroll
        for (int c = 0; c < 4; ++c) {
            const uint4* p = reinterpret_cast<const uint4*>(
                &kbuf[buf][rloc * ND + c * 32 + koff]);
            BfFrag f;
            f.q[0] = p[0];
            f.q[1] = p[2];
            b[c] = f.v;
        }

        v8f acc0 = {}, acc1 = {};
#pragma unroll
        for (int c = 0; c < 4; ++c) {
            acc0 = wmma_bf16(a[c],     b[c], acc0);
            acc1 = wmma_bf16(a[4 + c], b[c], acc1);
        }

        int col = lane & 15;
        int lr  = (lane >> 4) << 3;
#pragma unroll
        for (int v = 0; v < 8; ++v) {
            stile[wlocal][lr + v][col]      = acc0[v];
            stile[wlocal][16 + lr + v][col] = acc1[v];
        }
        __syncthreads();

#pragma unroll
        for (int j = 0; j < 16; ++j) {
            float sv = stile[wlocal][lane][j];
            int   id = kt * 16 + j;
            if (sv > minval) {
#pragma unroll
                for (int i = 0; i < NK; ++i)
                    if (i == minpos) { vals[i] = sv; ids[i] = id; }
                minval = vals[0]; minpos = 0;
#pragma unroll
                for (int i = 1; i < NK; ++i)
                    if (vals[i] < minval) { minval = vals[i]; minpos = i; }
            }
        }
    }

    int myrow = rowbase + lane;
#pragma unroll
    for (int j = 0; j < NK; ++j) topk[myrow * NK + j] = ids[j];
}

__global__ __launch_bounds__(256) void k_init(float* __restrict__ out,
                                              float* __restrict__ denom,
                                              int* __restrict__ m) {
    int i = blockIdx.x * 256 + threadIdx.x;
    if (i < NB * NHC) out[i] = 0.0f;
    if (i < NB * NH) { denom[i] = 0.0f; m[i] = (int)0x80000000; }
}

// Per-edge attention logits + segment max via ordered-int atomicMax.
__global__ __launch_bounds__(256) void k_score(const float* __restrict__ xl,
                                               const float* __restrict__ xr,
                                               const float* __restrict__ att,
                                               const int* __restrict__ topk,
                                               float* __restrict__ score,
                                               int* __restrict__ m) {
    int e = blockIdx.x * 256 + threadIdx.x;
    if (e >= NE) return;
    int src, dst; bool valid = true;
    if (e < NB * NK) { src = e >> 4; dst = topk[e]; valid = (src != dst); }
    else             { src = e - NB * NK; dst = src; }

    if (!valid) {
#pragma unroll
        for (int h = 0; h < NH; ++h) score[e * NH + h] = -1e9f;
        return;
    }

    const float4* pl = reinterpret_cast<const float4*>(xl + src * NHC);
    const float4* pr = reinterpret_cast<const float4*>(xr + dst * NHC);
    const float4* pa = reinterpret_cast<const float4*>(att);
#pragma unroll
    for (int h = 0; h < NH; ++h) {
        float s = 0.0f;
#pragma unroll
        for (int q = 0; q < NC / 4; ++q) {
            float4 l = pl[h * 16 + q], r = pr[h * 16 + q], aa = pa[h * 16 + q];
            float t;
            t = l.x + r.x; s += aa.x * (t > 0.0f ? t : NEG_SLOPE * t);
            t = l.y + r.y; s += aa.y * (t > 0.0f ? t : NEG_SLOPE * t);
            t = l.z + r.z; s += aa.z * (t > 0.0f ? t : NEG_SLOPE * t);
            t = l.w + r.w; s += aa.w * (t > 0.0f ? t : NEG_SLOPE * t);
        }
        score[e * NH + h] = s;
        atomicMax(&m[dst * NH + h], ford(s));
    }
}

// exp weights, denom accumulation, and weighted message scatter into out.
__global__ __launch_bounds__(256) void k_agg(const float* __restrict__ xl,
                                             const int* __restrict__ topk,
                                             const float* __restrict__ score,
                                             const int* __restrict__ m,
                                             float* __restrict__ denom,
                                             float* __restrict__ out) {
    int t = blockIdx.x * 256 + threadIdx.x;
    if (t >= NE * NH) return;
    int e = t >> 2, h = t & 3;
    float s = score[t];
    if (s < -1e8f) return;   // masked (removed self dup in top-k list)
    int src, dst;
    if (e < NB * NK) { src = e >> 4; dst = topk[e]; }
    else             { src = e - NB * NK; dst = src; }

    float w = __expf(s - fordinv(m[dst * NH + h]));
    atomicAdd(&denom[dst * NH + h], w);
    const float* msg = xl + src * NHC + h * NC;
    float* o         = out + dst * NHC + h * NC;
#pragma unroll 8
    for (int c = 0; c < NC; ++c)
        atomicAdd(&o[c], w * msg[c]);
}

__global__ __launch_bounds__(256) void k_final(float* __restrict__ out,
                                               const float* __restrict__ denom,
                                               const float* __restrict__ bias) {
    int i = blockIdx.x * 256 + threadIdx.x;
    if (i >= NB * NHC) return;
    int b = i >> 8, hc = i & 255, h = hc >> 6;
    out[i] = out[i] / denom[b * NH + h] + bias[hc];
}

// ---------- launch ----------

extern "C" void kernel_launch(void* const* d_in, const int* in_sizes, int n_in,
                              void* d_out, int out_size, void* d_ws, size_t ws_size,
                              hipStream_t stream) {
    const float* x   = (const float*)d_in[0];
    const float* W_l = (const float*)d_in[1];
    const float* b_l = (const float*)d_in[2];
    const float* W_r = (const float*)d_in[3];
    const float* b_r = (const float*)d_in[4];
    const float* att = (const float*)d_in[5];
    const float* bias= (const float*)d_in[6];
    float* out = (float*)d_out;

    char* ws = (char*)d_ws;
    size_t off = 0;
    unsigned short* xb  = (unsigned short*)(ws + off); off += (size_t)NB * ND * 2;     // 4 MB
    unsigned short* wlt = (unsigned short*)(ws + off); off += (size_t)NHC * ND * 2;    // 64 KB
    unsigned short* wrt = (unsigned short*)(ws + off); off += (size_t)NHC * ND * 2;    // 64 KB
    float* xl           = (float*)(ws + off);          off += (size_t)NB * NHC * 4;    // 16 MB
    float* xr           = (float*)(ws + off);          off += (size_t)NB * NHC * 4;    // 16 MB
    int*   topk         = (int*)(ws + off);            off += (size_t)NB * NK * 4;     // 1 MB
    float* score        = (float*)(ws + off);          off += (size_t)NE * NH * 4;     // ~4.25 MB
    int*   m            = (int*)(ws + off);            off += (size_t)NB * NH * 4;     // 256 KB
    float* denom        = (float*)(ws + off);          off += (size_t)NB * NH * 4;     // 256 KB

    k_cvt_x <<<(NB * ND + 255) / 256, 256, 0, stream>>>(x, xb);
    k_cvt_w <<<(ND * NHC + 255) / 256, 256, 0, stream>>>(W_l, W_r, wlt, wrt);
    k_linear<<<(NB / 16) * (NHC / 16) * 32 / 256, 256, 0, stream>>>(xb, wlt, wrt,
                                                                    b_l, b_r, xl, xr);
    k_init  <<<(NB * NHC + 255) / 256, 256, 0, stream>>>(out, denom, m);
    k_topk  <<<NB / 256, 256, 0, stream>>>(xb, topk);
    k_score <<<(NE + 255) / 256, 256, 0, stream>>>(xl, xr, att, topk, score, m);
    k_agg   <<<(NE * NH + 255) / 256, 256, 0, stream>>>(xl, topk, score, m, denom, out);
    k_final <<<(NB * NHC + 255) / 256, 256, 0, stream>>>(out, denom, bias);
}
